// SimpleFastAttention_87840671137905
// MI455X (gfx1250) — matrix operand order repaired and checked by reading, not compiled, not run on hardware
//
#include <hip/hip_runtime.h>
#include <hip/hip_bf16.h>
#include <math.h>

// Problem constants (from reference): B=8, T=2048, C=1024, H=16, HS=64
#define B_  8
#define T_  2048
#define C_  1024
#define H_  16
#define HS_ 64

typedef _Float16 f16;
typedef _Float16 v8h  __attribute__((ext_vector_type(8)));
typedef _Float16 v16h __attribute__((ext_vector_type(16)));
typedef float    v8f  __attribute__((ext_vector_type(8)));
typedef int      v4i_gcc __attribute__((vector_size(16)));  // async-LDS builtin param type

// CDNA5 async global->LDS copy path (guarded; falls back to VGPR round-trip)
#if __has_builtin(__builtin_amdgcn_global_load_async_to_lds_b128) && \
    __has_builtin(__builtin_amdgcn_s_wait_asynccnt)
#define USE_ASYNC_LDS 1
#else
#define USE_ASYNC_LDS 0
#endif

__device__ __forceinline__ v16h cat16(v8h lo, v8h hi) {
  return __builtin_shufflevector(lo, hi, 0,1,2,3,4,5,6,7,8,9,10,11,12,13,14,15);
}

// ---------------------------------------------------------------------------
// f32 -> f16 elementwise convert
// ---------------------------------------------------------------------------
__global__ void f32_to_f16_kernel(const float* __restrict__ in, f16* __restrict__ out, int n) {
  int i = blockIdx.x * blockDim.x + threadIdx.x;
  int stride = gridDim.x * blockDim.x;
  for (; i < n; i += stride) out[i] = (f16)in[i];
}

// ---------------------------------------------------------------------------
// Tiled LDS transpose with conversion to f16: out[c*rows + r] = in[r*cols + c]
// Used for: helper^T, M^T (so GEMM B operands are [N][K] row-major), and the
// faithful transpose(1,2).view(B,T,C) reshuffle (exactly [T,C]->[C,T]).
// ---------------------------------------------------------------------------
template <typename InT>
__global__ void transpose_to_f16_kernel(const InT* __restrict__ in, f16* __restrict__ out,
                                        int rows, int cols,
                                        long long inBatch, long long outBatch) {
  __shared__ float tile[32][33];
  in  += (long long)blockIdx.z * inBatch;
  out += (long long)blockIdx.z * outBatch;
  const int c0 = blockIdx.x * 32, r0 = blockIdx.y * 32;
  const int tx = threadIdx.x, ty = threadIdx.y;
#pragma unroll
  for (int i = 0; i < 4; ++i)
    tile[ty + 8 * i][tx] = (float)in[(long long)(r0 + ty + 8 * i) * cols + c0 + tx];
  __syncthreads();
#pragma unroll
  for (int i = 0; i < 4; ++i)
    out[(long long)(c0 + ty + 8 * i) * rows + r0 + tx] = (f16)tile[tx][ty + 8 * i];
}

// ---------------------------------------------------------------------------
// Generic batched WMMA GEMM with B supplied TRANSPOSED:
//   Cout[z] = alpha * (A[z] @ Bt[z]^T) (+ bias[col])
//   A : [M x K] row-major f16  (lda, batch stride aBatch)
//   Bt: [N x K] row-major f16  (ldb, batch stride bBatch; 0 for shared weights)
// Because Bt is [N][K], BOTH tiles stage into LDS as straight row-major b128
// copies (no scatter), and both fragment types are two contiguous 16-byte LDS
// loads matching the documented wave32 WMMA layouts.
// Block tile 128x128, K-step 32, double-buffered LDS; 8 wave32s, each wave a
// 32x64 sub-tile -> 8 v_wmma_f32_16x16x32_f16 per K-step.
// Requirements (all call sites): K%32==0, N%128==0, lda/ldb %8==0.
// ---------------------------------------------------------------------------
template <typename OutT>
__global__ __launch_bounds__(256) void gemm_bt_wmma_kernel(
    const f16* __restrict__ A, long long aBatch, int lda,
    const f16* __restrict__ Bt, long long bBatch, int ldb,
    OutT* __restrict__ Cout, long long cBatch, int ldc,
    int M, int N, int K, const float* __restrict__ bias, float alpha) {
  constexpr int BM = 128, BN = 128, BK = 32, LR = BK + 8;  // 80B rows, 16B aligned
  __shared__ __align__(16) f16 As[2][BM * LR];  // As[buf][m][k]
  __shared__ __align__(16) f16 Bs[2][BN * LR];  // Bs[buf][n][k]

  A    += (long long)blockIdx.z * aBatch;
  Bt   += (long long)blockIdx.z * bBatch;
  Cout += (long long)blockIdx.z * cBatch;

  const int tid = threadIdx.x;
  const int blockM = blockIdx.y * BM;
  const int blockN = blockIdx.x * BN;
  const int w = tid >> 5, lane = tid & 31;
  const int g = lane >> 4, ln = lane & 15;
  const int wm = (w & 3) * 32, wn = (w >> 2) * 64;
  const int kTiles = K / BK;

  // Stage one 128x32 A tile + 128x32 B tile: 2+2 b128 copies per thread.
  // OOB A-rows (M=64 head GEMM) are clamped; their results are never stored.
  auto stage = [&](int kt, int buf) {
    const int k0 = kt * BK;
#pragma unroll
    for (int it = 0; it < 2; ++it) {
      const int i = tid + it * 256;
      const int r = i >> 2;
      const int k = (i & 3) * 8;
      const int ga = (blockM + r < M) ? (blockM + r) : (M - 1);
      const f16* ap = A + (long long)ga * lda + k0 + k;
      const f16* bp = Bt + (long long)(blockN + r) * ldb + k0 + k;
#if USE_ASYNC_LDS
      __builtin_amdgcn_global_load_async_to_lds_b128(
          (v4i_gcc*)ap, (v4i_gcc*)&As[buf][r * LR + k], 0, 0);
      __builtin_amdgcn_global_load_async_to_lds_b128(
          (v4i_gcc*)bp, (v4i_gcc*)&Bs[buf][r * LR + k], 0, 0);
#else
      *(v8h*)&As[buf][r * LR + k] = *(const v8h*)ap;
      *(v8h*)&Bs[buf][r * LR + k] = *(const v8h*)bp;
      if (kt + 1 < kTiles) {                       // global_prefetch_b8 next tile
        __builtin_prefetch(ap + BK, 0, 3);
        __builtin_prefetch(bp + BK, 0, 3);
      }
#endif
    }
  };

  v8f zero = {};
  v8f acc[2][4];
#pragma unroll
  for (int mi = 0; mi < 2; ++mi)
#pragma unroll
    for (int ni = 0; ni < 4; ++ni) acc[mi][ni] = zero;

  stage(0, 0);
  for (int kt = 0; kt < kTiles; ++kt) {
    const int buf = kt & 1;
#if USE_ASYNC_LDS
    __builtin_amdgcn_s_wait_asynccnt(0);
#endif
    __syncthreads();
    if (kt + 1 < kTiles) stage(kt + 1, buf ^ 1);  // overlap with WMMA below

    v16h af[2], bf[4];
#pragma unroll
    for (int mi = 0; mi < 2; ++mi) {
      // A 16x32 layout: lanes 0-15 K={0..7,16..23}, lanes 16-31 K={8..15,24..31}
      const f16* p = &As[buf][(wm + mi * 16 + ln) * LR];
      af[mi] = cat16(*(const v8h*)(p + 8 * g), *(const v8h*)(p + 16 + 8 * g));
    }
#pragma unroll
    for (int ni = 0; ni < 4; ++ni) {
      // B 32x16 layout: lanes 0-15 K={0..7,16..23}, lanes 16-31 K={8..15,24..31}, N = lane%16
      const f16* p = &Bs[buf][(wn + ni * 16 + ln) * LR];
      bf[ni] = cat16(*(const v8h*)(p + 8 * g), *(const v8h*)(p + 16 + 8 * g));
    }
#pragma unroll
    for (int mi = 0; mi < 2; ++mi)
#pragma unroll
      for (int ni = 0; ni < 4; ++ni)
        acc[mi][ni] = __builtin_amdgcn_wmma_f32_16x16x32_f16(
            false, af[mi], false, bf[ni], (short)0, acc[mi][ni], false, false);
  }

  // Epilogue: C/D layout M = vgpr + 8*g, N = lane%16
#pragma unroll
  for (int mi = 0; mi < 2; ++mi)
#pragma unroll
    for (int ni = 0; ni < 4; ++ni)
#pragma unroll
      for (int i = 0; i < 8; ++i) {
        const int r = blockM + wm + mi * 16 + 8 * g + i;
        const int c = blockN + wn + ni * 16 + ln;
        if (r < M) {
          float v = acc[mi][ni][i] * alpha;
          if (bias) v += bias[c];
          Cout[(long long)r * ldc + c] = (OutT)v;
        }
      }
}

// ---------------------------------------------------------------------------
// Per-head attention logits: att[b,h,s,d] = sum_t P[t, h*64+s] * V[t, h*64+d]
// One block per (b,h); 64x64 output, K = T = 2048 in 32-steps, double-buffered.
// Both operands staged transposed (scatter) -- this kernel is ~1.5% of FLOPs.
// ---------------------------------------------------------------------------
__global__ __launch_bounds__(256) void attn_head_gemm_kernel(
    const f16* __restrict__ P, const f16* __restrict__ V, float* __restrict__ att) {
  constexpr int LR = 40;
  __shared__ __align__(16) f16 Us[2][HS_ * LR];  // Us[buf][s][t']
  __shared__ __align__(16) f16 Vt[2][HS_ * LR];  // Vt[buf][d][t']
  const int bh = blockIdx.x;
  const int b = bh / H_, h = bh % H_;
  const long long base = (long long)b * T_ * C_ + h * HS_;
  const int tid = threadIdx.x;
  const int w = tid >> 5, lane = tid & 31;
  const int g = lane >> 4, ln = lane & 15;
  const int wm = (w & 3) * 16, wn = (w >> 2) * 32;
  const int tp = tid >> 3, s0 = (tid & 7) * 8;  // stage: 32 t-rows x 8-half chunks
  constexpr int kTiles = T_ / 32;

  auto stage = [&](int kt, int buf) {
    const int k0 = kt * 32;
    v8h pv = *(const v8h*)(P + base + (long long)(k0 + tp) * C_ + s0);
    v8h vv = *(const v8h*)(V + base + (long long)(k0 + tp) * C_ + s0);
#pragma unroll
    for (int j = 0; j < 8; ++j) {
      Us[buf][(s0 + j) * LR + tp] = pv[j];
      Vt[buf][(s0 + j) * LR + tp] = vv[j];
    }
  };

  v8f zero = {};
  v8f acc[2] = {zero, zero};

  stage(0, 0);
  for (int kt = 0; kt < kTiles; ++kt) {
    const int buf = kt & 1;
    __syncthreads();
    if (kt + 1 < kTiles) stage(kt + 1, buf ^ 1);

    const f16* ap = &Us[buf][(wm + ln) * LR];
    v16h af = cat16(*(const v8h*)(ap + 8 * g), *(const v8h*)(ap + 16 + 8 * g));
#pragma unroll
    for (int ni = 0; ni < 2; ++ni) {
      const f16* bp = &Vt[buf][(wn + ni * 16 + ln) * LR];
      v16h bf = cat16(*(const v8h*)(bp + 8 * g), *(const v8h*)(bp + 16 + 8 * g));
      acc[ni] = __builtin_amdgcn_wmma_f32_16x16x32_f16(
          false, af, false, bf, (short)0, acc[ni], false, false);
    }
  }
  float* out = att + (long long)bh * HS_ * HS_;
#pragma unroll
  for (int ni = 0; ni < 2; ++ni)
#pragma unroll
    for (int i = 0; i < 8; ++i)
      out[(wm + 8 * g + i) * HS_ + wn + ni * 16 + ln] = acc[ni][i];
}

// ---------------------------------------------------------------------------
// Causal-masked softmax over 64-wide rows (scale = 1/sqrt(C) = 1/32) -> f16
// ---------------------------------------------------------------------------
__global__ void softmax_kernel(const float* __restrict__ att, f16* __restrict__ atth) {
  const int bh = blockIdx.x;
  const int r = threadIdx.x;
  const float* row = att + (long long)bh * HS_ * HS_ + r * HS_;
  f16* orow = atth + (long long)bh * HS_ * HS_ + r * HS_;
  const float scale = 0.03125f;
  float mx = -3.4e38f;
  for (int d = 0; d <= r; ++d) mx = fmaxf(mx, row[d] * scale);
  float sum = 0.0f;
  for (int d = 0; d <= r; ++d) sum += __expf(row[d] * scale - mx);
  const float inv = 1.0f / sum;
  for (int d = 0; d < HS_; ++d) {
    float v = (d <= r) ? __expf(row[d] * scale - mx) * inv : 0.0f;
    orow[d] = (f16)v;
  }
}

// ---------------------------------------------------------------------------
// Host-side orchestration (graph-capture safe: launches only)
// ---------------------------------------------------------------------------
extern "C" void kernel_launch(void* const* d_in, const int* in_sizes, int n_in,
                              void* d_out, int out_size, void* d_ws, size_t ws_size,
                              hipStream_t stream) {
  (void)in_sizes; (void)n_in; (void)out_size; (void)ws_size;
  const float* x   = (const float*)d_in[0];
  const float* wu  = (const float*)d_in[1];
  const float* wv  = (const float*)d_in[2];
  const float* wvb = (const float*)d_in[3];
  const float* hw  = (const float*)d_in[4];
  const float* cp  = (const float*)d_in[5];
  const float* cpb = (const float*)d_in[6];
  float* out = (float*)d_out;

  char* ws = (char*)d_ws;
  size_t off = 0;
  auto alloc = [&](size_t bytes) -> char* {
    char* p = ws + off;
    off += (bytes + 255) & ~(size_t)255;
    return p;
  };
  const long long TC = (long long)T_ * C_;
  f16* xh    = (f16*)alloc((size_t)B_ * TC * 2);        // x in f16
  f16* Pb    = (f16*)alloc((size_t)B_ * TC * 2);        // x @ wu^T ("u" as [T,C])
  f16* Vb    = (f16*)alloc((size_t)B_ * TC * 2);        // x @ wv^T + b
  f16* yh    = (f16*)alloc((size_t)B_ * TC * 2);        // x @ M
  f16* y2h   = (f16*)alloc((size_t)B_ * TC * 2);        // transpose-reshuffled y
  f16* Mb    = (f16*)alloc((size_t)B_ * C_ * C_ * 2);   // att @ helper, [B,C,C]
  f16* Mtb   = (f16*)alloc((size_t)B_ * C_ * C_ * 2);   // M^T per batch ([N][K])
  f16* wuH   = (f16*)alloc((size_t)C_ * C_ * 2);        // weights: native layout IS [N][K]
  f16* wvH   = (f16*)alloc((size_t)C_ * C_ * 2);
  f16* cpH   = (f16*)alloc((size_t)C_ * C_ * 2);
  f16* hwT   = (f16*)alloc((size_t)C_ * HS_ * 2);       // helper^T [C][HS]
  float* attf = (float*)alloc((size_t)B_ * H_ * HS_ * HS_ * 4);
  f16* atth  = (f16*)alloc((size_t)B_ * H_ * HS_ * HS_ * 2);

  const dim3 tb(32, 8);
  // Precision conversion; w_u/w_v/c_proj need NO transpose: for x @ w^T the
  // required Bt[n][k] = w[n][k] is exactly the native weight layout.
  f32_to_f16_kernel<<<4096, 256, 0, stream>>>(x, xh, B_ * T_ * C_);
  f32_to_f16_kernel<<<1024, 256, 0, stream>>>(wu, wuH, C_ * C_);
  f32_to_f16_kernel<<<1024, 256, 0, stream>>>(wv, wvH, C_ * C_);
  f32_to_f16_kernel<<<1024, 256, 0, stream>>>(cp, cpH, C_ * C_);
  transpose_to_f16_kernel<float><<<dim3(C_ / 32, HS_ / 32, 1), tb, 0, stream>>>(
      hw, hwT, HS_, C_, 0, 0);

  // u = x @ wu^T   (u[b,h,s,t] = Pb[b][t][h*64+s])
  gemm_bt_wmma_kernel<f16><<<dim3(C_ / 128, T_ / 128, B_), 256, 0, stream>>>(
      xh, TC, C_, wuH, 0, C_, Pb, TC, C_, T_, C_, C_, nullptr, 1.0f);
  // v = x @ wv^T + b
  gemm_bt_wmma_kernel<f16><<<dim3(C_ / 128, T_ / 128, B_), 256, 0, stream>>>(
      xh, TC, C_, wvH, 0, C_, Vb, TC, C_, T_, C_, C_, wvb, 1.0f);
  // att logits per head: u @ v over K = T
  attn_head_gemm_kernel<<<B_ * H_, 256, 0, stream>>>(Pb, Vb, attf);
  // scale + causal mask + softmax -> f16
  softmax_kernel<<<B_ * H_, HS_, 0, stream>>>(attf, atth);
  // M[b][(h*64+s)][c] = softatt[b,h,s,:] @ helper  (batched over b*h, M=64)
  gemm_bt_wmma_kernel<f16><<<dim3(C_ / 128, 1, B_ * H_), 256, 0, stream>>>(
      atth, (long long)HS_ * HS_, HS_, hwT, 0, HS_,
      Mb, (long long)HS_ * C_, C_, HS_, C_, HS_, nullptr, 1.0f);
  // M^T so y-GEMM's B operand is [N][K] row-major
  transpose_to_f16_kernel<f16><<<dim3(C_ / 32, C_ / 32, B_), tb, 0, stream>>>(
      Mb, Mtb, C_, C_, (long long)C_ * C_, (long long)C_ * C_);
  // y = x @ M[b]
  gemm_bt_wmma_kernel<f16><<<dim3(C_ / 128, T_ / 128, B_), 256, 0, stream>>>(
      xh, TC, C_, Mtb, (long long)C_ * C_, C_, yh, TC, C_, T_, C_, C_, nullptr, 1.0f);
  // faithful transpose(1,2).contiguous().view(B,T,C): [T,C] -> [C,T] per batch
  transpose_to_f16_kernel<f16><<<dim3(C_ / 32, T_ / 32, B_), tb, 0, stream>>>(
      yh, y2h, T_, C_, TC, TC);
  // out = y2 @ cp^T + b  (f32 straight to d_out; native cp layout is Bt)
  gemm_bt_wmma_kernel<float><<<dim3(C_ / 128, T_ / 128, B_), 256, 0, stream>>>(
      y2h, TC, C_, cpH, 0, C_, out, TC, C_, T_, C_, C_, cpb, 1.0f);
}
